// MultiBarrelModel_50208167690757
// MI455X (gfx1250) — compile-verified, matrix operand order checked by reading, not run on hardware
//
#include <hip/hip_runtime.h>
#include <hip/hip_bf16.h>
#include <cstdint>
#include <cstddef>

// ---------------------------------------------------------------------------
// Types for CDNA5 WMMA (wave32, v_wmma_f32_16x16x32_bf16)
// ---------------------------------------------------------------------------
typedef __attribute__((ext_vector_type(16))) __bf16 v16bf;
typedef __attribute__((ext_vector_type(8)))  float  v8f;

// ---------------------------------------------------------------------------
// Problem constants
// ---------------------------------------------------------------------------
#define BSZ   32
#define NNODE 39
#define TIN   307
#define T1LEN 124
#define T2LEN 64
#define NPOP  8
#define NROWS (T2LEN * BSZ * NNODE)   // 79872 gate rows
#define ROK   (128 * 11 * 39)         // 54912 readout K
#define ROM   (BSZ * 54)              // 1728 readout M rows
#define C2K   (64 * 21)               // 1344 conv2 K

__constant__ float COORD[39][2] = {
  {-6.f,0.f},{-5.3f,-3.f},{-5.3f,3.f},{-4.6f,-7.8f},{-4.6f,7.8f},{-3.5f,0.f},
  {-3.05f,-5.2f},{-3.05f,5.2f},{-3.1f,-1.75f},{-3.1f,1.75f},{-1.6f,-8.9f},
  {-1.75f,-3.f},{-1.6f,8.9f},{-1.75f,3.f},{-1.5f,0.f},{-0.7f,-1.3f},{-0.7f,1.3f},
  {0.f,-6.f},{0.f,6.f},{0.f,-3.5f},{0.f,0.f},{0.f,3.5f},{0.8f,-1.3f},{0.8f,1.3f},
  {1.6f,-8.9f},{1.6f,8.9f},{1.5f,0.f},{1.75f,3.f},{1.75f,-3.f},{3.05f,-5.2f},
  {3.1f,1.75f},{3.05f,5.2f},{3.1f,-1.75f},{3.6f,0.f},{4.6f,-7.8f},{4.6f,7.8f},
  {5.3f,-3.f},{5.3f,3.f},{6.f,0.f}};

// SRC[i] = populations feeding pop i (from ALLTOALL columns), flattened.
__constant__ int SRCc[37] = {0,1,2,3,4, 0,1,2,3,4, 1,2,3,5,6,7, 1,2,3,5,6,7,
                             0,4,6,7, 1,5,7, 2,3,4,5,6,7, 6,7};
__constant__ int OFFc[9]  = {0,5,10,16,22,26,29,35,37};

// ---------------------------------------------------------------------------
// Workspace layout (bytes).  conv1/conv2 region is reused by the packed
// readout weights (both dead by then).  Peak ~116 MB.
// ---------------------------------------------------------------------------
#define WS_CONV1   ((size_t)0)                        // 1248*64*124 f32 = 39,616,512
#define WS_CONV2   ((size_t)39616512)                 // 1248*64*64  f32 = 20,447,232
#define WS_ROPACK  ((size_t)0)                        // 512*54912 bf16 = 56,229,888 (alias)
#define WS_STAT1   ((size_t)60063744)                 // 128 f32
#define WS_STAT2   (WS_STAT1 + 512)
#define WS_XSBF    (WS_STAT2 + 512)                   // NROWS*64 bf16 = 10,223,616
#define WS_IDX     (WS_XSBF + 10223616)               // NROWS*6 int  = 1,916,928
#define WS_SPKBF   (WS_IDX + 1916928)                 // 8*32*39*32 bf16 = 638,976
#define WS_NBR     (WS_SPKBF + 638976)                // 39*4 int (pad 1024)
#define WS_BAR     (WS_NBR + 1024)                    // barrier counter (256)
#define WS_THW     (WS_BAR + 256)                     // 8*39*32*64 bf16 = 1,277,952
#define WS_POPW    (WS_THW + 1277952)                 // 37*39*32*32 bf16 = 2,955,264
#define WS_GW1T    (WS_POPW + 2955264)                // 256*64 bf16
#define WS_GW2T    (WS_GW1T + 32768)                  // 32*256 bf16
#define WS_AWT     (WS_GW2T + 16384)                  // 32*192 bf16
#define WS_RCONV   (WS_AWT + 12288)                   // 32*512*54 f32 = 3,538,944
#define WS_STAT3   (WS_RCONV + 3538944)               // 1024 f32
#define WS_POOL    (WS_STAT3 + 4096)                  // 32*512 f32 = 65,536
#define WS_C1BF    (WS_POOL + 65536)                  // 1248*64*124 bf16 = 19,808,256
#define WS_W2BF    (WS_C1BF + 19808256)               // 64*1344 bf16 = 172,032
#define WS_L56     (WS_W2BF + 172032)                 // 32*64*39*128 bf16 = 20,447,232

// ---------------------------------------------------------------------------
// Small helpers
// ---------------------------------------------------------------------------
__device__ __forceinline__ float gelu_f(float x) {
  return 0.5f * x * (1.0f + erff(x * 0.70710678118654752f));
}

__device__ __forceinline__ v8f wmma_bf(v16bf a, v16bf b, v8f c) {
  return __builtin_amdgcn_wmma_f32_16x16x32_bf16(false, a, false, b, (short)0, c,
                                                 false, false);
}

// CDNA5 async global->LDS copies (ASYNCcnt); per-lane addressing.
// Generic shared pointer low 32 bits == wave-relative LDS byte address.
__device__ __forceinline__ void async_b128(void* lds, const void* g) {
  asm volatile("global_load_async_to_lds_b128 %0, %1, off"
               :: "v"((unsigned)(size_t)lds), "v"(g) : "memory");
}
__device__ __forceinline__ void async_b64(void* lds, const void* g) {
  asm volatile("global_load_async_to_lds_b64 %0, %1, off"
               :: "v"((unsigned)(size_t)lds), "v"(g) : "memory");
}
__device__ __forceinline__ void wait_async0() {
  asm volatile("s_wait_asynccnt 0x0" ::: "memory");
}

// Load one 16x32 bf16 operand fragment from a row-major (rows x ldK) matrix.
__device__ __forceinline__ v16bf load_frag(const __bf16* base, int ldK) {
  int lane = threadIdx.x & 31;
  int r = lane & 15;
  int h = lane >> 4;
  union { v16bf v; unsigned u[8]; } fr;
  const __bf16* rowp = base + (size_t)r * ldK;
#pragma unroll
  for (int vv = 0; vv < 8; ++vv) {
    int klo = ((vv < 4) ? (2 * vv) : (8 + 2 * vv)) + 8 * h;
    fr.u[vv] = *(const unsigned*)(rowp + klo);
  }
  return fr.v;
}

// Store a 16x16 f32 C tile (VGPR r: lanes0-15 M=r, lanes16-31 M=r+8).
__device__ __forceinline__ void store_c(float* base, int ldN, v8f c) {
  int lane = threadIdx.x & 31;
  int ncol = lane & 15;
  int h = lane >> 4;
#pragma unroll
  for (int r = 0; r < 8; ++r) base[(size_t)(r + 8 * h) * ldN + ncol] = c[r];
}

// Software grid barrier (39 resident WGs; counter zeroed each launch).
__device__ __forceinline__ void gbar(unsigned* cnt, unsigned target) {
  __syncthreads();
  if (threadIdx.x == 0) {
    __threadfence();
    atomicAdd(cnt, 1u);
    while (atomicAdd(cnt, 0u) < target) __builtin_amdgcn_s_sleep(1);
    __threadfence();
  }
  __syncthreads();
}

// ---------------------------------------------------------------------------
// Stage 1a: conv1 (VALU fp32; tiny: 2-channel input, K=21, 0.8 GFLOP)
// ---------------------------------------------------------------------------
__global__ __launch_bounds__(256) void k_conv1(const float* __restrict__ x,
                                               const float* __restrict__ w1,
                                               const float* __restrict__ b1,
                                               float* __restrict__ out) {
  int e = blockIdx.x * 256 + threadIdx.x;
  if (e >= 1248 * 64 * T1LEN) return;
  int s = e / (64 * T1LEN);
  int r = e % (64 * T1LEN);
  int o = r / T1LEN, t1 = r % T1LEN;
  const float* xin = x + (size_t)s * 2 * TIN;
  float acc = b1[o];
#pragma unroll
  for (int ci = 0; ci < 2; ++ci)
#pragma unroll
    for (int k = 0; k < 21; ++k)
      acc += xin[ci * TIN + 2 * t1 + 3 * k] * w1[(o * 2 + ci) * 21 + k];
  out[e] = acc;
}

__global__ __launch_bounds__(256) void k_bnstats(const float* __restrict__ x,
                                                 float* __restrict__ stats,
                                                 int L, int S) {
  int ch = blockIdx.x, tid = threadIdx.x;
  __shared__ float rs[256], rq[256];
  int cnt = S * L;
  float s = 0.f, q = 0.f;
  for (int j = tid; j < cnt; j += 256) {
    float v = x[(size_t)(j / L) * (64 * L) + ch * L + (j % L)];
    s += v; q += v * v;
  }
  rs[tid] = s; rq[tid] = q; __syncthreads();
  for (int st = 128; st > 0; st >>= 1) {
    if (tid < st) { rs[tid] += rs[tid + st]; rq[tid] += rq[tid + st]; }
    __syncthreads();
  }
  if (tid == 0) {
    float m = rs[0] / cnt;
    stats[ch] = m;
    stats[64 + ch] = rq[0] / cnt - m * m;
  }
}

// normalize+gelu conv1 output; emit bf16 mirror for the conv2 WMMA GEMM
__global__ __launch_bounds__(256) void k_norm1(float* __restrict__ buf,
                                               const float* __restrict__ stats,
                                               const float* __restrict__ g,
                                               const float* __restrict__ be,
                                               __bf16* __restrict__ bfm) {
  int e = blockIdx.x * 256 + threadIdx.x;
  if (e >= 1248 * 64 * T1LEN) return;
  int o = (e / T1LEN) % 64;
  float m = stats[o], v = stats[64 + o];
  float xn = gelu_f((buf[e] - m) * rsqrtf(v + 1e-5f) * g[o] + be[o]);
  buf[e] = xn;
  bfm[e] = (__bf16)xn;
}

// ---------------------------------------------------------------------------
// Stage 1b: conv2 as implicit GEMM (M=79872, N=64, K=1344) via bf16 WMMA.
// comp_w2 (O,I,K) flattens directly to the row-major (N x 1344) operand.
// ---------------------------------------------------------------------------
__global__ __launch_bounds__(128) void k_conv2w(const __bf16* __restrict__ c1bf,
                                                const __bf16* __restrict__ w2bf,
                                                const float* __restrict__ b2,
                                                float* __restrict__ out) {
  __shared__ __bf16 As[16 * C2K];
  int g = blockIdx.x, tid = threadIdx.x, w = tid >> 5;
  int s = g >> 2, t2b = (g & 3) * 16;
  for (int e = tid; e < 16 * C2K; e += 128) {
    int r = e / C2K, k = e % C2K;
    int i = k / 21, kk = k % 21;
    As[e] = c1bf[((size_t)s * 64 + i) * T1LEN + (t2b + r) + 3 * kk];
  }
  __syncthreads();
  int ob = w * 16;
  v8f acc = {};
#pragma unroll 2
  for (int kb = 0; kb < C2K / 32; ++kb)
    acc = wmma_bf(load_frag(As + kb * 32, C2K),
                  load_frag(w2bf + (size_t)ob * C2K + kb * 32, C2K), acc);
  int lane = tid & 31, h = lane >> 4, ncl = lane & 15;
#pragma unroll
  for (int r = 0; r < 8; ++r) {
    int t2 = t2b + r + 8 * h, oc = ob + ncl;
    out[((size_t)s * 64 + oc) * T2LEN + t2] = acc[r] + b2[oc];
  }
}

__global__ __launch_bounds__(256) void k_xs(const float* __restrict__ buf,
                                            const float* __restrict__ stats,
                                            const float* __restrict__ g,
                                            const float* __restrict__ be,
                                            __bf16* __restrict__ xsbf) {
  int e = blockIdx.x * 256 + threadIdx.x;
  if (e >= 1248 * 64 * T2LEN) return;
  int s = e / (64 * T2LEN);
  int r = e % (64 * T2LEN);
  int o = r >> 6, t = r & 63;
  int b = s / NNODE, n = s % NNODE;
  float m = stats[o], v = stats[64 + o];
  float xn = (buf[e] - m) * rsqrtf(v + 1e-5f) * g[o] + be[o];
  xsbf[(((size_t)t * BSZ + b) * NNODE + n) * 64 + o] = (__bf16)xn;
}

// ---------------------------------------------------------------------------
// Weight packing / k-NN table
// ---------------------------------------------------------------------------
__global__ __launch_bounds__(64) void k_nbr(int* __restrict__ nbr) {
  int n = threadIdx.x;
  if (n >= NNODE) return;
  float d2[NNODE];
  for (int j = 0; j < NNODE; ++j) {
    float dx = COORD[n][0] - COORD[j][0], dy = COORD[n][1] - COORD[j][1];
    d2[j] = dx * dx + dy * dy;
  }
  unsigned long long used = 1ull << n;
  for (int m = 0; m < 4; ++m) {
    float best = 3.4e38f; int bi = 0;
    for (int j = 0; j < NNODE; ++j) {
      if ((used >> j) & 1ull) continue;
      if (d2[j] < best) { best = d2[j]; bi = j; }
    }
    used |= 1ull << bi;
    nbr[n * 4 + m] = bi;
  }
}

__global__ __launch_bounds__(256) void k_pack_w(const float* thW, const float* popW,
                                                const float* gw1, const float* gw2,
                                                const float* aw, const float* w2,
                                                __bf16* thWb, __bf16* popWb,
                                                __bf16* gw1T, __bf16* gw2T,
                                                __bf16* awT, __bf16* w2b) {
  int e = blockIdx.x * 256 + threadIdx.x;
  if (e < 638976) { thWb[e] = (__bf16)thW[e]; return; }
  e -= 638976;
  if (e < 1477632) { popWb[e] = (__bf16)popW[e]; return; }
  e -= 1477632;
  if (e < 16384) { int j = e / 64, k = e % 64; gw1T[e] = (__bf16)gw1[k * 256 + j]; return; }
  e -= 16384;
  if (e < 8192) { int o = e / 256, j = e % 256; gw2T[e] = (__bf16)gw2[j * 32 + o]; return; }
  e -= 8192;
  if (e < 6144) { int o = e / 192, k = e % 192; awT[e] = (__bf16)aw[k * 32 + o]; return; }
  e -= 6144;
  if (e < 64 * C2K) { w2b[e] = (__bf16)w2[e]; return; }
}

__global__ __launch_bounds__(256) void k_pack_ro(const float* __restrict__ row,
                                                 __bf16* __restrict__ out) {
  int e = blockIdx.x * 256 + threadIdx.x;
  if (e >= 512 * ROK) return;
  int o = e / ROK, k = e % ROK;
  int kh = k / (39 * 128);
  int rem = k - kh * 39 * 128;
  int n = rem >> 7, c = rem & 127;
  out[e] = (__bf16)row[(((size_t)o * 128 + c) * 11 + kh) * 39 + n];
}

// ---------------------------------------------------------------------------
// Gate precompute: h1=gelu(LN(x@gw1+gb1)), g=h1@gw2+gb2, write top-6 indices.
// ---------------------------------------------------------------------------
__global__ __launch_bounds__(256) void k_gate(const __bf16* __restrict__ xsbf,
                                              const __bf16* __restrict__ gw1T,
                                              const __bf16* __restrict__ gw2T,
                                              const float* __restrict__ gb1,
                                              const float* __restrict__ glng,
                                              const float* __restrict__ glnb,
                                              const float* __restrict__ gb2,
                                              int* __restrict__ idxb) {
  __shared__ float s_h1[32 * 256];
  __shared__ float s_g[32 * 32];
  __shared__ __bf16 s_h1g[32 * 256];
  int tid = threadIdx.x, w = tid >> 5;
  size_t R0 = (size_t)blockIdx.x * 32;

  {
    int mi = w & 1;
    const __bf16* A = xsbf + (R0 + mi * 16) * 64;
    v16bf a0 = load_frag(A, 64);
    v16bf a1 = load_frag(A + 32, 64);
    for (int q = 0; q < 4; ++q) {
      int nj = (w >> 1) * 4 + q;
      const __bf16* B = gw1T + (size_t)nj * 16 * 64;
      v8f acc = {};
      acc = wmma_bf(a0, load_frag(B, 64), acc);
      acc = wmma_bf(a1, load_frag(B + 32, 64), acc);
      store_c(s_h1 + mi * 16 * 256 + nj * 16, 256, acc);
    }
  }
  __syncthreads();
  if (tid < 32) {
    float m = 0.f, q = 0.f;
    for (int j = 0; j < 256; ++j) { float v = s_h1[tid * 256 + j] + gb1[j]; m += v; q += v * v; }
    m *= (1.f / 256.f);
    float rstd = rsqrtf(q * (1.f / 256.f) - m * m + 1e-5f);
    for (int j = 0; j < 256; ++j) {
      float v = s_h1[tid * 256 + j] + gb1[j];
      s_h1g[tid * 256 + j] = (__bf16)gelu_f((v - m) * rstd * glng[j] + glnb[j]);
    }
  }
  __syncthreads();
  if (w < 4) {
    int mi = w >> 1, nj = w & 1;
    const __bf16* A = s_h1g + mi * 16 * 256;
    const __bf16* B = gw2T + (size_t)nj * 16 * 256;
    v8f acc = {};
#pragma unroll
    for (int kb = 0; kb < 8; ++kb)
      acc = wmma_bf(load_frag(A + kb * 32, 256), load_frag(B + kb * 32, 256), acc);
    store_c(s_g + mi * 16 * 32 + nj * 16, 32, acc);
  }
  __syncthreads();
  if (tid < 32) {
    float vals[32];
    for (int o = 0; o < 32; ++o) vals[o] = s_g[tid * 32 + o] + gb2[o];
    unsigned used = 0;
    for (int j = 0; j < 6; ++j) {
      float best = -3.4e38f; int bi = 0;
      for (int o = 0; o < 32; ++o)
        if (!((used >> o) & 1u) && vals[o] > best) { best = vals[o]; bi = o; }
      used |= 1u << bi;
      idxb[(R0 + tid) * 6 + j] = bi;
    }
  }
}

// ---------------------------------------------------------------------------
// Persistent recurrent kernel: 39 WGs (one per node), 8 waves, ~237 KB LDS.
// x_t and the gated neighbor modules are staged via async global->LDS copies.
// ---------------------------------------------------------------------------
#define REC_SMEM 237056

__global__ __launch_bounds__(256) void k_recurrent(
    const __bf16* __restrict__ xsbf, const int* __restrict__ idxb,
    __bf16* __restrict__ spkg, const int* __restrict__ nbrg, unsigned* cnt,
    const __bf16* __restrict__ thWb, const __bf16* __restrict__ popWb,
    const __bf16* __restrict__ awTb, const float* __restrict__ th_b,
    const float* __restrict__ pop_b, const float* __restrict__ tau_adp,
    const float* __restrict__ tau_m, const float* __restrict__ mems0,
    const float* __restrict__ spikes0, const float* __restrict__ b0,
    const float* __restrict__ abv, const float* __restrict__ alng,
    const float* __restrict__ alnb, float* __restrict__ hstate,
    float* __restrict__ mstate, __bf16* __restrict__ l56bf) {
  extern __shared__ char smem[];
  __bf16* s_thW  = (__bf16*)smem;            // 8*32*64
  __bf16* s_popW = s_thW + 16384;            // 37*32*32
  __bf16* s_awT  = s_popW + 37888;           // 32*192
  __bf16* s_xt   = s_awT + 6144;             // 32*64
  __bf16* s_sel  = s_xt + 2048;              // 32*192
  __bf16* s_spk  = s_sel + 6144;             // 8*32*32
  float* s_mem   = (float*)(s_spk + 8192);   // 8*32*32
  float* s_bb    = s_mem + 8192;
  float* s_cur   = s_bb + 8192;              // 32*32
  float* s_agg   = s_cur + 1024;
  float* s_stage = s_agg + 1024;
  float* s_thb   = s_stage + 1024;           // 8*32
  float* s_pbs   = s_thb + 256;
  float* s_al    = s_pbs + 256;
  float* s_ro    = s_al + 256;
  float* s_ab    = s_ro + 256;               // 32 each
  float* s_lng   = s_ab + 32;
  float* s_lnb   = s_lng + 32;
  float* s_am    = s_lnb + 32;
  float* s_ar    = s_am + 32;
  int*   s_idx   = (int*)(s_ar + 32);        // 32*6
  int*   s_nbr   = s_idx + 192;              // 4

  const int n = blockIdx.x;
  const int tid = threadIdx.x;
  const int w = tid >> 5;

  // ---- init ----
  for (int e = tid; e < 8 * 2048; e += 256) {
    int i = e >> 11;
    s_thW[e] = thWb[((size_t)i * NNODE + n) * 2048 + (e & 2047)];
  }
  for (int e = tid; e < 37 * 1024; e += 256) {
    int k = e >> 10;
    s_popW[e] = popWb[((size_t)k * NNODE + n) * 1024 + (e & 1023)];
  }
  for (int e = tid; e < 6144; e += 256) s_awT[e] = awTb[e];
  if (tid < 256) {
    int i = tid >> 5, o = tid & 31;
    s_thb[tid] = th_b[(i * NNODE + n) * 32 + o];
    float ps = 0.f;
    for (int k = OFFc[i]; k < OFFc[i + 1]; ++k) ps += pop_b[(k * NNODE + n) * 32 + o];
    s_pbs[tid] = ps;
    s_al[tid] = expf(-1.f / tau_m[(i * NNODE + n) * 32 + o]);
    s_ro[tid] = expf(-1.f / tau_adp[(i * NNODE + n) * 32 + o]);
  }
  if (tid < 32) { s_ab[tid] = abv[tid]; s_lng[tid] = alng[tid]; s_lnb[tid] = alnb[tid]; }
  if (tid < 4) s_nbr[tid] = nbrg[n * 4 + tid];
  for (int e = tid; e < 8192; e += 256) {
    int i = e >> 10, b = (e >> 5) & 31, o = e & 31;
    size_t gi = (((size_t)i * BSZ + b) * NNODE + n) * 32 + o;
    s_mem[e] = mems0[gi];
    float sp = spikes0[gi];
    s_spk[e] = (__bf16)sp;
    s_bb[e] = b0[i * NNODE + n];
    spkg[gi] = (__bf16)sp;
  }
  unsigned ep = 1;
  gbar(cnt, NNODE * ep); ++ep;

  // ---- scan ----
  for (int t = 0; t < T2LEN; ++t) {
    // async-stage x_t tile: 32 rows x 128B (per-lane b128)
    {
      int b = tid >> 3, seg = tid & 7;
      async_b128(s_xt + b * 64 + seg * 8,
                 xsbf + (((size_t)t * BSZ + b) * NNODE + n) * 64 + seg * 8);
    }
    for (int e = tid; e < 192; e += 256)
      s_idx[e] = idxb[(((size_t)t * BSZ + (e / 6)) * NNODE + n) * 6 + (e % 6)];
    wait_async0();
    __syncthreads();

    for (int i = 0; i < NPOP; ++i) {
      // async-gather selected neighbor modules (192 x 64B, per-lane b64)
#pragma unroll
      for (int q = 0; q < 6; ++q) {
        int u = q * 256 + tid;
        int b = u / 48, r4 = (u % 48) * 4;
        int j = r4 >> 5, f0 = r4 & 31;
        int qq = s_idx[b * 6 + j];
        int nb = s_nbr[qq >> 3], p = qq & 7;
        async_b64(s_sel + u * 4,
                  spkg + (((size_t)p * BSZ + b) * NNODE + nb) * 32 + f0);
      }
      wait_async0();
      __syncthreads();

      if (w < 4) {  // cur = x_t@thW^T + sum spk@popW^T
        int mi = w >> 1, nj = w & 1;
        v8f acc = {};
        const __bf16* A = s_xt + mi * 16 * 64;
        const __bf16* B = s_thW + i * 2048 + nj * 16 * 64;
        acc = wmma_bf(load_frag(A, 64), load_frag(B, 64), acc);
        acc = wmma_bf(load_frag(A + 32, 64), load_frag(B + 32, 64), acc);
        for (int kk = OFFc[i]; kk < OFFc[i + 1]; ++kk) {
          int sp = SRCc[kk];
          acc = wmma_bf(load_frag(s_spk + sp * 1024 + mi * 512, 32),
                        load_frag(s_popW + kk * 1024 + nj * 512, 32), acc);
        }
        store_c(s_cur + mi * 512 + nj * 16, 32, acc);
      } else {      // agg = Sel(32x192) @ aw
        int u = w - 4, mi = u >> 1, nj = u & 1;
        v8f acc = {};
        const __bf16* A = s_sel + mi * 16 * 192;
        const __bf16* B = s_awT + nj * 16 * 192;
#pragma unroll
        for (int kb = 0; kb < 6; ++kb)
          acc = wmma_bf(load_frag(A + kb * 32, 192), load_frag(B + kb * 32, 192), acc);
        store_c(s_agg + mi * 512 + nj * 16, 32, acc);
      }
      __syncthreads();

      if (tid < 32) {  // LN stats of (agg + ab) per batch row
        float m = 0.f, q = 0.f;
        for (int o = 0; o < 32; ++o) { float v = s_agg[tid * 32 + o] + s_ab[o]; m += v; q += v * v; }
        m *= (1.f / 32.f);
        s_am[tid] = m;
        s_ar[tid] = rsqrtf(q * (1.f / 32.f) - m * m + 1e-5f);
      }
      __syncthreads();

      for (int e = tid; e < 1024; e += 256) {   // mem_update
        int b = e >> 5, o = e & 31;
        float aggv = ((s_agg[e] + s_ab[o]) - s_am[b]) * s_ar[b] * s_lng[o] + s_lnb[o];
        float cur = s_cur[e] + s_thb[i * 32 + o] + s_pbs[i * 32 + o] + aggv;
        int si = i * 1024 + e;
        float mem = s_mem[si], bb = s_bb[si], spko = (float)s_spk[si];
        float al = s_al[i * 32 + o], ro = s_ro[i * 32 + o];
        bb = ro * bb + (1.f - ro) * spko;
        float Bth = 0.02f + 1.8f * bb;
        mem = mem * al + (1.f - al) * cur - Bth * spko;
        float spn = (mem - Bth) > 0.f ? 1.f : 0.f;
        s_mem[si] = mem; s_bb[si] = bb;
        s_spk[si] = (__bf16)spn; s_stage[e] = spn;
        size_t ho = ((((size_t)b * T2LEN + t) * NNODE) + n) * 256 + i * 32 + o;
        hstate[ho] = spn;
        mstate[ho] = mem + spn * Bth;
        if (i >= 4)
          l56bf[((((size_t)b * T2LEN + t) * NNODE) + n) * 128 + (i - 4) * 32 + o] =
              (__bf16)spn;
      }
      gbar(cnt, NNODE * ep); ++ep;   // all nodes done reading old spk[i]
      for (int e = tid; e < 1024; e += 256) {
        int b = e >> 5, o = e & 31;
        spkg[(((size_t)i * BSZ + b) * NNODE + n) * 32 + o] = (__bf16)s_stage[e];
      }
      gbar(cnt, NNODE * ep); ++ep;   // commit visible before next phase
    }
  }
}

// ---------------------------------------------------------------------------
// Readout conv as implicit GEMM: M=1728, N=512, K=54912, bf16 WMMA.
// A tiles async-staged from the bf16 l56 spike buffer (contiguous 128B rows).
// ---------------------------------------------------------------------------
__global__ __launch_bounds__(256) void k_readout(const __bf16* __restrict__ l56bf,
                                                 const __bf16* __restrict__ ropack,
                                                 float* __restrict__ convout) {
  __shared__ __bf16 As[16 * 64];
  int g = blockIdx.x, tid = threadIdx.x, w = tid >> 5;
  v8f acc[4] = {{}, {}, {}, {}};
  for (int kb = 0; kb < ROK / 64; ++kb) {
    if (w < 4) {  // 16 rows x 128B via per-lane async b128 (4 wave-issues)
      int u = tid;                   // 0..127
      int r = u >> 3, seg = u & 7;
      int row = g * 16 + r, b = row / 54, to = row % 54;
      int kh = (kb * 64) / 4992, rem = kb * 64 - kh * 4992;
      int n = rem >> 7, c0 = rem & 127;
      async_b128(As + r * 64 + seg * 8,
                 l56bf + (((size_t)b * T2LEN + (to + kh)) * NNODE + n) * 128 +
                     c0 + seg * 8);
    }
    wait_async0();
    __syncthreads();
    v16bf a0 = load_frag(As, 64);
    v16bf a1 = load_frag(As + 32, 64);
#pragma unroll
    for (int q = 0; q < 4; ++q) {
      const __bf16* B = ropack + (size_t)((w * 4 + q) * 16) * ROK + kb * 64;
      __builtin_prefetch(B + 64, 0, 1);
      acc[q] = wmma_bf(a0, load_frag(B, ROK), acc[q]);
      acc[q] = wmma_bf(a1, load_frag(B + 32, ROK), acc[q]);
    }
    __syncthreads();
  }
  int lane = tid & 31, h = lane >> 4, ncl = lane & 15;
#pragma unroll
  for (int q = 0; q < 4; ++q)
#pragma unroll
    for (int r = 0; r < 8; ++r) {
      int row = g * 16 + r + 8 * h, b = row / 54, to = row % 54;
      int oc = (w * 4 + q) * 16 + ncl;
      convout[((size_t)b * 512 + oc) * 54 + to] = acc[q][r];
    }
}

__global__ __launch_bounds__(256) void k_bnstats3(const float* __restrict__ x,
                                                  const float* __restrict__ bias,
                                                  float* __restrict__ stats) {
  int ch = blockIdx.x, tid = threadIdx.x;
  __shared__ float rs[256], rq[256];
  float s = 0.f, q = 0.f;
  for (int j = tid; j < 32 * 54; j += 256) {
    float v = x[((size_t)(j / 54) * 512 + ch) * 54 + (j % 54)] + bias[ch];
    s += v; q += v * v;
  }
  rs[tid] = s; rq[tid] = q; __syncthreads();
  for (int st = 128; st > 0; st >>= 1) {
    if (tid < st) { rs[tid] += rs[tid + st]; rq[tid] += rq[tid + st]; }
    __syncthreads();
  }
  if (tid == 0) {
    float m = rs[0] / (32.f * 54.f);
    stats[ch] = m;
    stats[512 + ch] = rq[0] / (32.f * 54.f) - m * m;
  }
}

__global__ __launch_bounds__(256) void k_pool(const float* __restrict__ x,
                                              const float* __restrict__ bias,
                                              const float* __restrict__ stats,
                                              const float* __restrict__ g,
                                              const float* __restrict__ be,
                                              float* __restrict__ pooled) {
  int id = blockIdx.x * 256 + threadIdx.x;
  if (id >= 32 * 512) return;
  int b = id / 512, ch = id % 512;
  float m = stats[ch], rstd = rsqrtf(stats[512 + ch] + 1e-5f);
  float s = 0.f;
  for (int t = 0; t < 54; ++t) {
    float v = x[((size_t)b * 512 + ch) * 54 + t] + bias[ch];
    s += gelu_f((v - m) * rstd * g[ch] + be[ch]);
  }
  pooled[id] = s / 54.f;
}

__global__ __launch_bounds__(256) void k_classifier(const float* __restrict__ pooled,
                                                    const float* __restrict__ cw1,
                                                    const float* __restrict__ cb1,
                                                    const float* __restrict__ clng,
                                                    const float* __restrict__ clnb,
                                                    const float* __restrict__ cw2,
                                                    const float* __restrict__ cb2,
                                                    float* __restrict__ out) {
  __shared__ float s_p[512], s_h[256], s_r[256], rs[256], rq[256], s_l[36], red[2];
  int b = blockIdx.x, tid = threadIdx.x;
  for (int j = tid; j < 512; j += 256) s_p[j] = pooled[b * 512 + j];
  __syncthreads();
  float h = cb1[tid];
  for (int k = 0; k < 512; ++k) h += s_p[k] * cw1[k * 256 + tid];
  s_h[tid] = h; rs[tid] = h; rq[tid] = h * h;
  __syncthreads();
  for (int st = 128; st > 0; st >>= 1) {
    if (tid < st) { rs[tid] += rs[tid + st]; rq[tid] += rq[tid + st]; }
    __syncthreads();
  }
  float m = rs[0] / 256.f;
  float rstd = rsqrtf(rq[0] / 256.f - m * m + 1e-5f);
  s_r[tid] = gelu_f((s_h[tid] - m) * rstd * clng[tid] + clnb[tid]);
  __syncthreads();
  if (tid < 36) {
    float l = cb2[tid];
    for (int k = 0; k < 256; ++k) l += s_r[k] * cw2[k * 36 + tid];
    s_l[tid] = l;
  }
  __syncthreads();
  if (tid == 0) {
    float mx = -3.4e38f;
    for (int j = 0; j < 36; ++j) mx = fmaxf(mx, s_l[j]);
    float sum = 0.f;
    for (int j = 0; j < 36; ++j) sum += expf(s_l[j] - mx);
    red[0] = mx; red[1] = sum;
  }
  __syncthreads();
  if (tid < 36) out[b * 36 + tid] = expf(s_l[tid] - red[0]) / red[1];
}

// ---------------------------------------------------------------------------
// Host-side orchestration
// ---------------------------------------------------------------------------
extern "C" void kernel_launch(void* const* d_in, const int* in_sizes, int n_in,
                              void* d_out, int out_size, void* d_ws, size_t ws_size,
                              hipStream_t stream) {
  (void)in_sizes; (void)n_in; (void)out_size; (void)ws_size;
  const float* inp     = (const float*)d_in[0];
  const float* mems0   = (const float*)d_in[1];
  const float* spikes0 = (const float*)d_in[2];
  const float* b0      = (const float*)d_in[3];
  const float* comp_w1 = (const float*)d_in[4];
  const float* comp_b1 = (const float*)d_in[5];
  const float* comp_g1 = (const float*)d_in[6];
  const float* comp_be1= (const float*)d_in[7];
  const float* comp_w2 = (const float*)d_in[8];
  const float* comp_b2 = (const float*)d_in[9];
  const float* comp_g2 = (const float*)d_in[10];
  const float* comp_be2= (const float*)d_in[11];
  const float* th_W    = (const float*)d_in[12];
  const float* th_b    = (const float*)d_in[13];
  const float* pop_W   = (const float*)d_in[14];
  const float* pop_b   = (const float*)d_in[15];
  const float* tau_adp = (const float*)d_in[16];
  const float* tau_m   = (const float*)d_in[17];
  const float* gw1     = (const float*)d_in[18];
  const float* gb1     = (const float*)d_in[19];
  const float* gln_g   = (const float*)d_in[20];
  const float* gln_b   = (const float*)d_in[21];
  const float* gw2     = (const float*)d_in[22];
  const float* gb2     = (const float*)d_in[23];
  const float* aw      = (const float*)d_in[24];
  const float* ab      = (const float*)d_in[25];
  const float* aln_g   = (const float*)d_in[26];
  const float* aln_b   = (const float*)d_in[27];
  const float* ro_w    = (const float*)d_in[28];
  const float* ro_b    = (const float*)d_in[29];
  const float* ro_g    = (const float*)d_in[30];
  const float* ro_be   = (const float*)d_in[31];
  const float* cw1     = (const float*)d_in[32];
  const float* cb1     = (const float*)d_in[33];
  const float* cln_g   = (const float*)d_in[34];
  const float* cln_b   = (const float*)d_in[35];
  const float* cw2     = (const float*)d_in[36];
  const float* cb2     = (const float*)d_in[37];

  char* ws = (char*)d_ws;
  float*  conv1  = (float*)(ws + WS_CONV1);
  float*  conv2  = (float*)(ws + WS_CONV2);
  __bf16* ropack = (__bf16*)(ws + WS_ROPACK);
  float*  stat1  = (float*)(ws + WS_STAT1);
  float*  stat2  = (float*)(ws + WS_STAT2);
  __bf16* xsbf   = (__bf16*)(ws + WS_XSBF);
  int*    idxb   = (int*)(ws + WS_IDX);
  __bf16* spkg   = (__bf16*)(ws + WS_SPKBF);
  int*    nbrg   = (int*)(ws + WS_NBR);
  unsigned* bar  = (unsigned*)(ws + WS_BAR);
  __bf16* thWb   = (__bf16*)(ws + WS_THW);
  __bf16* popWb  = (__bf16*)(ws + WS_POPW);
  __bf16* gw1T   = (__bf16*)(ws + WS_GW1T);
  __bf16* gw2T   = (__bf16*)(ws + WS_GW2T);
  __bf16* awT    = (__bf16*)(ws + WS_AWT);
  float*  rconv  = (float*)(ws + WS_RCONV);
  float*  stat3  = (float*)(ws + WS_STAT3);
  float*  pooled = (float*)(ws + WS_POOL);
  __bf16* c1bf   = (__bf16*)(ws + WS_C1BF);
  __bf16* w2bf   = (__bf16*)(ws + WS_W2BF);
  __bf16* l56bf  = (__bf16*)(ws + WS_L56);

  float* out_soft = (float*)d_out;
  float* hstate = out_soft + 32 * 36;
  float* mstate = hstate + (size_t)BSZ * T2LEN * NNODE * 256;

  (void)hipMemsetAsync(bar, 0, 256, stream);

  // tables + packed weights (independent of activations)
  k_nbr<<<1, 64, 0, stream>>>(nbrg);
  k_pack_w<<<(2233344 + 255) / 256, 256, 0, stream>>>(th_W, pop_W, gw1, gw2, aw,
                                                      comp_w2, thWb, popWb, gw1T,
                                                      gw2T, awT, w2bf);

  // compressor
  k_conv1<<<(1248 * 64 * T1LEN + 255) / 256, 256, 0, stream>>>(inp, comp_w1, comp_b1, conv1);
  k_bnstats<<<64, 256, 0, stream>>>(conv1, stat1, T1LEN, 1248);
  k_norm1<<<(1248 * 64 * T1LEN + 255) / 256, 256, 0, stream>>>(conv1, stat1, comp_g1,
                                                               comp_be1, c1bf);
  k_conv2w<<<NROWS / 16, 128, 0, stream>>>(c1bf, w2bf, comp_b2, conv2);
  k_bnstats<<<64, 256, 0, stream>>>(conv2, stat2, T2LEN, 1248);
  k_xs<<<(1248 * 64 * T2LEN + 255) / 256, 256, 0, stream>>>(conv2, stat2, comp_g2, comp_be2, xsbf);

  // gate precompute (WMMA)
  k_gate<<<NROWS / 32, 256, 0, stream>>>(xsbf, gw1T, gw2T, gb1, gln_g, gln_b, gb2, idxb);

  // pack readout weights (reuses conv1/conv2 region, both dead now)
  k_pack_ro<<<(512 * ROK + 255) / 256, 256, 0, stream>>>(ro_w, ropack);

  // persistent recurrent scan (WMMA + async-LDS + grid barriers)
  k_recurrent<<<NNODE, 256, REC_SMEM, stream>>>(
      xsbf, idxb, spkg, nbrg, bar, thWb, popWb, awT, th_b, pop_b, tau_adp, tau_m,
      mems0, spikes0, b0, ab, aln_g, aln_b, hstate, mstate, l56bf);

  // readout conv as bf16 WMMA GEMM with async A staging
  k_readout<<<ROM / 16, 256, 0, stream>>>(l56bf, ropack, rconv);
  k_bnstats3<<<512, 256, 0, stream>>>(rconv, ro_b, stat3);
  k_pool<<<(32 * 512 + 255) / 256, 256, 0, stream>>>(rconv, ro_b, stat3, ro_g, ro_be, pooled);
  k_classifier<<<32, 256, 0, stream>>>(pooled, cw1, cb1, cln_g, cln_b, cw2, cb2, out_soft);
}
